// BahdanauAttention_61727269978752
// MI455X (gfx1250) — compile-verified, matrix operand order checked by reading, not run on hardware
//
#include <hip/hip_runtime.h>
#include <hip/hip_bf16.h>
#include <math.h>

typedef __attribute__((ext_vector_type(16))) __bf16 v16bf;
typedef __attribute__((ext_vector_type(4)))  __bf16 v4bf;
typedef __attribute__((ext_vector_type(8)))  float  v8f;

#define B_  32
#define S_  2048
#define D_  1024
#define U_  1024

// Native CDNA5 tanh (V_TANH_F32, TRANS32 op) instead of branchy libm tanhf.
__device__ __forceinline__ float fast_tanh(float x) {
#if __has_builtin(__builtin_amdgcn_tanhf)
  return __builtin_amdgcn_tanhf(x);
#else
  float y;
  // v_nop covers the TRANS-op result-use hazard the compiler can't see.
  asm volatile("v_tanh_f32 %0, %1\n\tv_nop" : "=v"(y) : "v"(x));
  return y;
#endif
}

// ---------------------------------------------------------------------------
// Kernel 1: pack W1 (fp32, [D,U] row-major) into bf16 per-lane B-fragment
// layout for v_wmma_f32_16x16x32_bf16.
// Packed index: ((c*32 + kb)*32 + lane)*16 + e
//   lane<16:  N=lane,     K in {0..7, 16..23} of the 32-K block
//   lane>=16: N=lane-16,  K in {8..15, 24..31}
// ---------------------------------------------------------------------------
__global__ __launch_bounds__(256) void pack_w1_kernel(
    const float* __restrict__ W1, __bf16* __restrict__ packed) {
  int idx  = blockIdx.x * 256 + threadIdx.x;      // 0 .. 1M-1
  int e    = idx & 15;
  int lane = (idx >> 4) & 31;
  int kb   = (idx >> 9) & 31;
  int c    = idx >> 14;
  int v    = e >> 1, pos = e & 1;
  int hi   = lane >> 4;
  int n    = lane & 15;
  int k_local = (v < 4 ? 2 * v + pos : 16 + 2 * (v - 4) + pos) + (hi ? 8 : 0);
  int k = kb * 32 + k_local;
  int u = c * 16 + n;
  packed[idx] = (__bf16)W1[k * U_ + u];
}

// ---------------------------------------------------------------------------
// Kernel 2: qb[b,u] = query[b,:] @ W2[:,u] + b2[u] + b1[u]   (bias folded)
// ---------------------------------------------------------------------------
__global__ __launch_bounds__(1024) void proj_q_kernel(
    const float* __restrict__ query, const float* __restrict__ W2,
    const float* __restrict__ b1, const float* __restrict__ b2,
    float* __restrict__ qb) {
  int b = blockIdx.x;
  int u = threadIdx.x;                       // coalesced over u
  const float* q = query + b * D_;
  float acc = b1[u] + b2[u];
  #pragma unroll 8
  for (int d = 0; d < D_; ++d)
    acc += q[d] * W2[d * U_ + u];
  qb[b * U_ + u] = acc;
}

// ---------------------------------------------------------------------------
// Kernel 3: fused  score[b,s] = tanh(values[b,s,:]@W1 + b1 + qb[b,:]) . V + bV
// One WG per (b, 32-row s-tile). 8 waves; each wave owns 8 u-chunks of 16 and
// drives TWO 16x16 accumulators (rows 0..15 and 16..31) off one shared B
// fragment, halving packed-W1 (L2) traffic per FLOP.
// ---------------------------------------------------------------------------
union FragU {
  uint4 u[2];
  v16bf h;
};

__global__ __launch_bounds__(256) void score_kernel(
    const float*  __restrict__ values,
    const __bf16* __restrict__ packedW1,
    const float*  __restrict__ qb,
    const float*  __restrict__ Vvec,
    const float*  __restrict__ bV,
    float*        __restrict__ scores) {
  __shared__ __bf16 sA[32 * D_];   // 64 KB: 32 s-rows x 1024 k, bf16
  __shared__ float  sQB[U_];       // 4 KB
  __shared__ float  sV[U_];        // 4 KB
  __shared__ float  sScore[32];

  const int b  = blockIdx.y;
  const int s0 = blockIdx.x * 32;
  const int t  = threadIdx.x;

  // stage A tile (fp32 -> bf16), vectorized: float4 load -> 4x bf16 -> b64 store
  const float* vbase = values + ((size_t)b * S_ + s0) * D_;
  #pragma unroll
  for (int i = 0; i < 32; ++i) {
    int idx = (i * 256 + t) * 4;
    float4 f = *(const float4*)&vbase[idx];
    v4bf h;
    h[0] = (__bf16)f.x; h[1] = (__bf16)f.y;
    h[2] = (__bf16)f.z; h[3] = (__bf16)f.w;
    *(v4bf*)&sA[idx] = h;
  }
  #pragma unroll
  for (int i = 0; i < 4; ++i) {
    sQB[i * 256 + t] = qb[b * U_ + i * 256 + t];
    sV[i * 256 + t]  = Vvec[i * 256 + t];
  }
  if (t < 32) sScore[t] = 0.0f;
  __syncthreads();

  const int wave = t >> 5;
  const int lane = t & 31;
  const int hi   = lane >> 4;       // which K-half this lane holds
  const int row  = lane & 15;       // A row (within 16) / C column index

  float sc0[8], sc1[8];
  #pragma unroll
  for (int r = 0; r < 8; ++r) { sc0[r] = 0.0f; sc1[r] = 0.0f; }

  const int abase0 = row * D_;            // rows 0..15
  const int abase1 = (row + 16) * D_;     // rows 16..31

  for (int cc = 0; cc < 8; ++cc) {
    const int c = cc * 8 + wave;    // u-chunk 0..63
    v8f acc0 = {};
    v8f acc1 = {};

    #pragma unroll 2
    for (int kb = 0; kb < 32; ++kb) {
      // A fragments from LDS: two 16-byte runs per lane, per row-half
      FragU a0, a1;
      a0.u[0] = *(const uint4*)&sA[abase0 + kb * 32 + hi * 8];
      a0.u[1] = *(const uint4*)&sA[abase0 + kb * 32 + 16 + hi * 8];
      a1.u[0] = *(const uint4*)&sA[abase1 + kb * 32 + hi * 8];
      a1.u[1] = *(const uint4*)&sA[abase1 + kb * 32 + 16 + hi * 8];
      // B fragment from packed global: 32 contiguous bytes per lane (L2-hot)
      const v16bf bfrag =
          *(const v16bf*)(packedW1 + (((size_t)c * 32 + kb) * 32 + lane) * 16);
      acc0 = __builtin_amdgcn_wmma_f32_16x16x32_bf16(
          false, a0.h, false, bfrag, (short)0, acc0, false, false);
      acc1 = __builtin_amdgcn_wmma_f32_16x16x32_bf16(
          false, a1.h, false, bfrag, (short)0, acc1, false, false);
    }

    // epilogue: C element (M = r + hi*8 [+16], N = row), u = c*16 + N
    const int   u   = c * 16 + row;
    const float qbu = sQB[u];
    const float vu  = sV[u];
    #pragma unroll
    for (int r = 0; r < 8; ++r) {
      sc0[r] += fast_tanh(acc0[r] + qbu) * vu;
      sc1[r] += fast_tanh(acc1[r] + qbu) * vu;
    }
  }

  // reduce over N (16 lanes within each half of the wave)
  #pragma unroll
  for (int r = 0; r < 8; ++r) {
    float v0 = sc0[r], v1 = sc1[r];
    for (int off = 1; off < 16; off <<= 1) {
      v0 += __shfl_xor(v0, off, 32);
      v1 += __shfl_xor(v1, off, 32);
    }
    sc0[r] = v0; sc1[r] = v1;
  }
  // lane 0 holds rows {0..7, 16..23} partials, lane 16 rows {8..15, 24..31}
  if ((lane & 15) == 0) {
    #pragma unroll
    for (int r = 0; r < 8; ++r) {
      atomicAdd(&sScore[r + hi * 8], sc0[r]);        // ds_add_f32
      atomicAdd(&sScore[16 + r + hi * 8], sc1[r]);
    }
  }
  __syncthreads();
  if (t < 32)
    scores[(size_t)b * S_ + s0 + t] = sScore[t] + bV[0];
}

// ---------------------------------------------------------------------------
// Kernel 4: softmax over S per batch; writes weights (ws) + attn output
// ---------------------------------------------------------------------------
__global__ __launch_bounds__(256) void softmax_kernel(
    const float* __restrict__ scores,
    float* __restrict__ weights, float* __restrict__ out_attn) {
  __shared__ float red[256];
  const int b = blockIdx.x;
  const int t = threadIdx.x;

  float loc[8];
  float mx = -INFINITY;
  #pragma unroll
  for (int i = 0; i < 8; ++i) {
    loc[i] = scores[b * S_ + i * 256 + t];
    mx = fmaxf(mx, loc[i]);
  }
  red[t] = mx;
  __syncthreads();
  for (int s = 128; s > 0; s >>= 1) {
    if (t < s) red[t] = fmaxf(red[t], red[t + s]);
    __syncthreads();
  }
  mx = red[0];
  __syncthreads();

  float sum = 0.0f;
  #pragma unroll
  for (int i = 0; i < 8; ++i) {
    loc[i] = __expf(loc[i] - mx);
    sum += loc[i];
  }
  red[t] = sum;
  __syncthreads();
  for (int s = 128; s > 0; s >>= 1) {
    if (t < s) red[t] += red[t + s];
    __syncthreads();
  }
  const float inv = 1.0f / red[0];
  #pragma unroll
  for (int i = 0; i < 8; ++i) {
    const float w = loc[i] * inv;
    weights[b * S_ + i * 256 + t]  = w;
    out_attn[b * S_ + i * 256 + t] = w;
  }
}

// ---------------------------------------------------------------------------
// Kernel 5: context[b,d] = sum_s w[b,s] * values[b,s,d]   (streaming)
// ---------------------------------------------------------------------------
__global__ __launch_bounds__(256) void context_kernel(
    const float* __restrict__ values, const float* __restrict__ weights,
    float* __restrict__ ctx) {
  const int b = blockIdx.y;
  const int d = blockIdx.x * 256 + threadIdx.x;
  const float* vb = values + (size_t)b * S_ * D_ + d;
  const float* wb = weights + b * S_;
  float acc = 0.0f;
  for (int s = 0; s < S_; s += 8) {
    #pragma unroll
    for (int j = 0; j < 8; ++j)
      acc += wb[s + j] * vb[(size_t)(s + j) * D_];
  }
  ctx[b * D_ + d] = acc;
}

// ---------------------------------------------------------------------------
extern "C" void kernel_launch(void* const* d_in, const int* in_sizes, int n_in,
                              void* d_out, int out_size, void* d_ws, size_t ws_size,
                              hipStream_t stream) {
  const float* query  = (const float*)d_in[0];
  const float* values = (const float*)d_in[1];
  const float* W1     = (const float*)d_in[2];
  const float* b1     = (const float*)d_in[3];
  const float* W2     = (const float*)d_in[4];
  const float* b2     = (const float*)d_in[5];
  const float* Vvec   = (const float*)d_in[6];
  const float* bV     = (const float*)d_in[7];

  char* ws = (char*)d_ws;
  __bf16* packedW1 = (__bf16*)(ws);                              // 2 MB
  float*  qb       = (float*)(ws + 2u * 1024u * 1024u);          // 128 KB
  float*  scores   = (float*)(ws + 2u * 1024u * 1024u + 131072u);// 256 KB
  float*  weights  = (float*)(ws + 2u * 1024u * 1024u + 131072u + 262144u);

  float* ctx  = (float*)d_out;            // [B, D]  = 32768 floats
  float* attn = (float*)d_out + B_ * D_;  // [B, S, 1] = 65536 floats

  pack_w1_kernel <<<4096, 256, 0, stream>>>(W1, packedW1);
  proj_q_kernel  <<<B_, 1024, 0, stream>>>(query, W2, b1, b2, qb);
  score_kernel   <<<dim3(S_ / 32, B_), 256, 0, stream>>>(
      values, packedW1, qb, Vvec, bV, scores);
  softmax_kernel <<<B_, 256, 0, stream>>>(scores, weights, attn);
  context_kernel <<<dim3(D_ / 256, B_), 256, 0, stream>>>(values, weights, ctx);
}
